// DualAttention_7146825581055
// MI455X (gfx1250) — matrix operand order repaired and checked by reading, not compiled, not run on hardware
//
#include <hip/hip_runtime.h>
#include <math.h>

// ---------------------------------------------------------------------------
// Problem constants (from reference): B=4, N=1024, QN=512, C=1024, H=16, D=64
// ---------------------------------------------------------------------------
#define PB 4
#define PN 1024
#define PQN 512
#define PC 1024
#define PH 16
#define PD 64

typedef __bf16 bf16;
typedef __attribute__((ext_vector_type(16))) __bf16 v16bf;
typedef __attribute__((ext_vector_type(8)))  __bf16 v8bf;
typedef __attribute__((ext_vector_type(4)))  __bf16 v4bf;
typedef __attribute__((ext_vector_type(8)))  float  v8f;

// ---------------------------------------------------------------------------
// CDNA5 async global->LDS copy (ASYNCcnt-tracked). Guarded so the file still
// compiles (with a synchronous fallback) if this toolchain lacks the builtins.
// Probe-discovered signature: (v4i addrspace(1)*, v4i addrspace(3)*, imm, imm)
// ---------------------------------------------------------------------------
#if defined(__has_builtin)
#if __has_builtin(__builtin_amdgcn_global_load_async_to_lds_b128)
#define USE_ASYNC_LDS 1
#endif
#endif

#ifdef USE_ASYNC_LDS
typedef int gv4i __attribute__((vector_size(16)));
typedef __attribute__((address_space(1))) gv4i* gas1_v4i;
typedef __attribute__((address_space(3))) gv4i* gas3_v4i;

__device__ __forceinline__ void async_cp16(const bf16* g, bf16* l) {
  __builtin_amdgcn_global_load_async_to_lds_b128(
      (gas1_v4i)(void*)const_cast<bf16*>(g),
      (gas3_v4i)(void*)l,
      /*offset=*/0, /*cpol=*/0);
}
#if __has_builtin(__builtin_amdgcn_s_wait_asynccnt)
#define WAITA(n) __builtin_amdgcn_s_wait_asynccnt(n)
#else
#define WAITA(n) asm volatile("s_wait_asynccnt %0" ::"i"(n) : "memory")
#endif
#else
#define WAITA(n)
#endif

// Build a 16-element bf16 fragment from two 8-element (16-byte) chunks.
__device__ __forceinline__ v16bf ld16_2(const bf16* p0, const bf16* p1) {
  v8bf a = *(const v8bf*)p0;
  v8bf b = *(const v8bf*)p1;
  v16bf r;
#pragma unroll
  for (int i = 0; i < 8; ++i) { r[i] = a[i]; r[i + 8] = b[i]; }
  return r;
}

// ---------------------------------------------------------------------------
// f32 -> bf16 bulk conversion (memory-bound prologue, ~4us of 23.3TB/s HBM)
// ---------------------------------------------------------------------------
__global__ __launch_bounds__(256)
void cvt_f32_bf16(const float* __restrict__ in, bf16* __restrict__ out,
                  int n4) {
  int i = blockIdx.x * blockDim.x + threadIdx.x;
  if (i < n4) {
    float4 v = ((const float4*)in)[i];
    v4bf c;
    c[0] = (bf16)v.x; c[1] = (bf16)v.y; c[2] = (bf16)v.z; c[3] = (bf16)v.w;
    ((v4bf*)out)[i] = c;
  }
}

// o = a + b (bf16, 8-wide)
__global__ __launch_bounds__(256)
void add_bf16_vec(const bf16* __restrict__ a, const bf16* __restrict__ b,
                  bf16* __restrict__ o, int n8) {
  int i = blockIdx.x * blockDim.x + threadIdx.x;
  if (i < n8) {
    v8bf va = ((const v8bf*)a)[i];
    v8bf vb = ((const v8bf*)b)[i];
    v8bf vo;
#pragma unroll
    for (int j = 0; j < 8; ++j) vo[j] = (bf16)((float)va[j] + (float)vb[j]);
    ((v8bf*)o)[i] = vo;
  }
}

// ---------------------------------------------------------------------------
// WMMA GEMM: OUT(M,Nc) = A(M,K) * W(Nc,K)^T, A/W bf16 row-major.
//   mode==1: OUT f32 row-major + bias (final projections, writes d_out)
//   mode==0: scatter epilogue into per-head bf16 tensors:
//            which = n / C selects p0/p1/p2;
//            which == vtIdx  -> [B,H,D,S] (transposed V for attention B-frags)
//            else            -> [B,H,S,D]
// Block: 256 threads (8 waves), tile 128x128x32, double-buffered async LDS.
// Wave tile 32x64 = 2x4 WMMA tiles.
// ---------------------------------------------------------------------------
#define GSTR 40  // 32 + 8 pad; rows stay 16B aligned (80B)

__device__ __forceinline__ void stage_tile(const bf16* __restrict__ A,
                                           const bf16* __restrict__ W, int K,
                                           int mBase, int nBase, int kt,
                                           int tid, bf16* As, bf16* Ws) {
  const int lrow = tid >> 2;         // 0..63
  const int lcol = (tid & 3) << 3;   // 0,8,16,24
#pragma unroll
  for (int p = 0; p < 2; ++p) {
    int row = lrow + p * 64;
    const bf16* ga = A + (size_t)(mBase + row) * K + (size_t)kt * 32 + lcol;
    const bf16* gw = W + (size_t)(nBase + row) * K + (size_t)kt * 32 + lcol;
#ifdef USE_ASYNC_LDS
    async_cp16(ga, &As[row * GSTR + lcol]);
    async_cp16(gw, &Ws[row * GSTR + lcol]);
#else
    *(v8bf*)&As[row * GSTR + lcol] = *(const v8bf*)ga;
    *(v8bf*)&Ws[row * GSTR + lcol] = *(const v8bf*)gw;
#endif
  }
}

__global__ __launch_bounds__(256)
void gemm_bf16_wmma(const bf16* __restrict__ A, const bf16* __restrict__ W,
                    int M, int Nc, int K, int mode, float* __restrict__ outF,
                    const float* __restrict__ bias, bf16* __restrict__ p0,
                    bf16* __restrict__ p1, bf16* __restrict__ p2, int vtIdx,
                    int S) {
  __shared__ bf16 As[2][128 * GSTR];
  __shared__ bf16 Ws[2][128 * GSTR];

  const int tid   = threadIdx.x;
  const int lane  = tid & 31;
  const int wave  = tid >> 5;
  const int wm    = wave & 3;   // wave M position (0..3)
  const int wn    = wave >> 2;  // wave N position (0..1)
  const int mBase = blockIdx.x * 128;
  const int nBase = blockIdx.y * 128;

  v8f acc[2][4];
#pragma unroll
  for (int mi = 0; mi < 2; ++mi)
#pragma unroll
    for (int ni = 0; ni < 4; ++ni)
#pragma unroll
      for (int r = 0; r < 8; ++r) acc[mi][ni][r] = 0.0f;

  const int nk = K >> 5;
  stage_tile(A, W, K, mBase, nBase, 0, tid, As[0], Ws[0]);

  for (int kt = 0; kt < nk; ++kt) {
    const int cur = kt & 1;
    if (kt + 1 < nk) {
      stage_tile(A, W, K, mBase, nBase, kt + 1, tid, As[cur ^ 1], Ws[cur ^ 1]);
      WAITA(4);  // 4 async ops outstanding for next tile; this tile complete
    } else {
      WAITA(0);
    }
    __syncthreads();

    const int k0 = (lane >> 4) << 3;   // A-frag K origin per lane half
    const int c0 = (lane >> 4) << 3;   // B-frag K origin per lane half
    v16bf af[2], bfm[4];
#pragma unroll
    for (int mi = 0; mi < 2; ++mi) {
      const bf16* pa = &As[cur][(32 * wm + 16 * mi + (lane & 15)) * GSTR + k0];
      af[mi] = ld16_2(pa, pa + 16);
    }
#pragma unroll
    for (int ni = 0; ni < 4; ++ni) {
      const bf16* pb = &Ws[cur][(64 * wn + 16 * ni + (lane & 15)) * GSTR + c0];
      bfm[ni] = ld16_2(pb, pb + 16);
    }
#pragma unroll
    for (int mi = 0; mi < 2; ++mi)
#pragma unroll
      for (int ni = 0; ni < 4; ++ni)
        acc[mi][ni] = __builtin_amdgcn_wmma_f32_16x16x32_bf16(
            false, af[mi], false, bfm[ni], (short)0, acc[mi][ni], false, false);
    __syncthreads();
  }

  // Epilogue
#pragma unroll
  for (int mi = 0; mi < 2; ++mi)
#pragma unroll
    for (int ni = 0; ni < 4; ++ni)
#pragma unroll
      for (int r = 0; r < 8; ++r) {
        int m = mBase + 32 * wm + 16 * mi + ((lane >> 4) << 3) + r;
        int n = nBase + 64 * wn + 16 * ni + (lane & 15);
        float v = acc[mi][ni][r];
        if (mode == 1) {
          outF[(size_t)m * Nc + n] = v + bias[n];
        } else {
          int b = m / S, sTok = m % S;
          int which = n >> 10;  // n / C
          int c = n & (PC - 1);
          int hh = c >> 6, dd = c & (PD - 1);
          bf16* p = (which == 0) ? p0 : ((which == 1) ? p1 : p2);
          size_t idx;
          if (which == vtIdx)
            idx = (((size_t)(b * PH + hh)) * PD + dd) * S + sTok;  // [B,H,D,S]
          else
            idx = (((size_t)(b * PH + hh)) * S + sTok) * PD + dd;  // [B,H,S,D]
          p[idx] = (bf16)v;
        }
      }
}

// ---------------------------------------------------------------------------
// Flash attention (WMMA), D=64.
//   Q: [B,H,Sq,D] bf16   K: [B,H,Sk,D] bf16   Vt: [B,H,D,Sk] bf16
//   O: [B,Sq,H,D] bf16;  O = tanh(gate[h]) * softmax(Q K^T / 8) V
// Block: 128 threads (4 waves); each wave owns 16 query rows; 64-key tiles,
// double-buffered async LDS staging for K/V.
// ---------------------------------------------------------------------------
#define FSTR 72

__device__ __forceinline__ void stage_kv(const bf16* __restrict__ Kbh,
                                         const bf16* __restrict__ Vbh, int Sk,
                                         int kt, int tid, bf16* Kt, bf16* Vt) {
  const int row  = tid >> 1;
  const int half = tid & 1;
  const bf16* kg = Kbh + (size_t)(kt * 64 + row) * PD + half * 32;
  const bf16* vg = Vbh + (size_t)row * Sk + (size_t)kt * 64 + half * 32;
#pragma unroll
  for (int q4 = 0; q4 < 4; ++q4) {
#ifdef USE_ASYNC_LDS
    async_cp16(kg + q4 * 8, &Kt[row * FSTR + half * 32 + q4 * 8]);
    async_cp16(vg + q4 * 8, &Vt[row * FSTR + half * 32 + q4 * 8]);
#else
    *(v8bf*)&Kt[row * FSTR + half * 32 + q4 * 8] = *(const v8bf*)(kg + q4 * 8);
    *(v8bf*)&Vt[row * FSTR + half * 32 + q4 * 8] = *(const v8bf*)(vg + q4 * 8);
#endif
  }
}

__global__ __launch_bounds__(128)
void flash_attn_wmma(const bf16* __restrict__ Q, const bf16* __restrict__ Kb,
                     const bf16* __restrict__ Vt, bf16* __restrict__ O, int Sq,
                     int Sk, const float* __restrict__ gatep) {
  __shared__ bf16 Kt[2][64 * FSTR];
  __shared__ bf16 Vtile[2][64 * FSTR];
  __shared__ bf16 Ps[4 * 16 * FSTR];  // per-wave P staging (D -> A layout)

  const int tid  = threadIdx.x;
  const int lane = tid & 31;
  const int wave = tid >> 5;
  const int bh   = blockIdx.y;  // b*H + h
  const int h    = bh & (PH - 1);
  const int b    = bh >> 4;
  const int mBase = blockIdx.x * 64 + wave * 16;

  const bf16* Qb  = Q + (size_t)bh * Sq * PD;
  const bf16* Kbh = Kb + (size_t)bh * Sk * PD;
  const bf16* Vbh = Vt + (size_t)bh * PD * Sk;

  // Q fragments for this wave's 16 rows, pre-scaled by 1/sqrt(D)=0.125 (exact)
  const int mrow = mBase + (lane & 15);
  v16bf aq[2];
#pragma unroll
  for (int s = 0; s < 2; ++s) {
    const int k0 = 32 * s + ((lane >> 4) << 3);
    const bf16* qp = Qb + (size_t)mrow * PD + k0;
    v16bf t = ld16_2(qp, qp + 16);
#pragma unroll
    for (int i = 0; i < 16; ++i) t[i] = (bf16)((float)t[i] * 0.125f);
    aq[s] = t;
  }

  float rmax[8], rsum[8];
  v8f oacc[4];
#pragma unroll
  for (int r = 0; r < 8; ++r) { rmax[r] = -3.0e38f; rsum[r] = 0.0f; }
#pragma unroll
  for (int jd = 0; jd < 4; ++jd)
#pragma unroll
    for (int r = 0; r < 8; ++r) oacc[jd][r] = 0.0f;

  const int nkt = Sk >> 6;
  stage_kv(Kbh, Vbh, Sk, 0, tid, Kt[0], Vtile[0]);

  for (int kt = 0; kt < nkt; ++kt) {
    const int cur = kt & 1;
    if (kt + 1 < nkt) {
      stage_kv(Kbh, Vbh, Sk, kt + 1, tid, Kt[cur ^ 1], Vtile[cur ^ 1]);
      WAITA(8);
    } else {
      WAITA(0);
    }
    __syncthreads();

    // S = Q K^T  (16 x 64 per wave; 4 n-blocks x 2 k-steps)
    v8f sf[4];
#pragma unroll
    for (int ni = 0; ni < 4; ++ni) {
      v8f cz;
#pragma unroll
      for (int r = 0; r < 8; ++r) cz[r] = 0.0f;
      const bf16* pb =
          &Kt[cur][(16 * ni + (lane & 15)) * FSTR + ((lane >> 4) << 3)];
      v16bf b0 = ld16_2(pb, pb + 16);
      v16bf b1 = ld16_2(pb + 32, pb + 48);
      cz = __builtin_amdgcn_wmma_f32_16x16x32_bf16(false, aq[0], false, b0,
                                                   (short)0, cz, false, false);
      cz = __builtin_amdgcn_wmma_f32_16x16x32_bf16(false, aq[1], false, b1,
                                                   (short)0, cz, false, false);
      sf[ni] = cz;
    }

    // Online softmax: row stats live across the 16 lanes of each half-wave
    float corr[8];
#pragma unroll
    for (int r = 0; r < 8; ++r) {
      float tm = fmaxf(fmaxf(sf[0][r], sf[1][r]), fmaxf(sf[2][r], sf[3][r]));
#pragma unroll
      for (int mk = 1; mk <= 8; mk <<= 1) tm = fmaxf(tm, __shfl_xor(tm, mk, 32));
      float nm = fmaxf(rmax[r], tm);
      corr[r] = __expf(rmax[r] - nm);
      rmax[r] = nm;
      float ps = 0.0f;
#pragma unroll
      for (int ni = 0; ni < 4; ++ni) {
        float p = __expf(sf[ni][r] - nm);
        sf[ni][r] = p;
        ps += p;
      }
#pragma unroll
      for (int mk = 1; mk <= 8; mk <<= 1) ps += __shfl_xor(ps, mk, 32);
      rsum[r] = rsum[r] * corr[r] + ps;
    }
#pragma unroll
    for (int jd = 0; jd < 4; ++jd)
#pragma unroll
      for (int r = 0; r < 8; ++r) oacc[jd][r] *= corr[r];

    // Redistribute P through per-wave LDS (C/D layout -> A layout)
    bf16* Pw = &Ps[wave * 16 * FSTR];
#pragma unroll
    for (int ni = 0; ni < 4; ++ni)
#pragma unroll
      for (int r = 0; r < 8; ++r)
        Pw[(((lane >> 4) << 3) + r) * FSTR + 16 * ni + (lane & 15)] =
            (bf16)sf[ni][r];

    // O += P V  (keys are the reduction dim; Vt columns are contiguous)
#pragma unroll
    for (int s = 0; s < 2; ++s) {
      const int k0 = 32 * s + ((lane >> 4) << 3);
      const bf16* pp = &Pw[(lane & 15) * FSTR + k0];
      v16bf ap = ld16_2(pp, pp + 16);
#pragma unroll
      for (int jd = 0; jd < 4; ++jd) {
        const bf16* pv = &Vtile[cur][(16 * jd + (lane & 15)) * FSTR + 32 * s +
                                     ((lane >> 4) << 3)];
        v16bf bv = ld16_2(pv, pv + 16);
        oacc[jd] = __builtin_amdgcn_wmma_f32_16x16x32_bf16(
            false, ap, false, bv, (short)0, oacc[jd], false, false);
      }
    }
    __syncthreads();
  }

  const float g = gatep ? tanhf(gatep[h]) : 1.0f;
#pragma unroll
  for (int jd = 0; jd < 4; ++jd)
#pragma unroll
    for (int r = 0; r < 8; ++r) {
      int m = mBase + ((lane >> 4) << 3) + r;
      int d = 16 * jd + (lane & 15);
      float val = g * oacc[jd][r] / rsum[r];
      size_t idx = (((size_t)b * Sq + m) * PH + h) * PD + d;  // [B,Sq,H,D]
      O[idx] = (bf16)val;
    }
}

// ---------------------------------------------------------------------------
// Host launcher
// ---------------------------------------------------------------------------
extern "C" void kernel_launch(void* const* d_in, const int* in_sizes, int n_in,
                              void* d_out, int out_size, void* d_ws,
                              size_t ws_size, hipStream_t stream) {
  const float* x       = (const float*)d_in[0];
  const float* query   = (const float*)d_in[1];
  const float* w_qkv   = (const float*)d_in[2];
  const float* w_kv    = (const float*)d_in[3];
  const float* w_qlin  = (const float*)d_in[4];
  const float* gate    = (const float*)d_in[5];
  const float* w_proj  = (const float*)d_in[6];
  const float* b_proj  = (const float*)d_in[7];
  const float* w_qproj = (const float*)d_in[8];
  const float* b_qproj = (const float*)d_in[9];

  float* x_out = (float*)d_out;
  float* q_out = x_out + (size_t)PB * PN * PC;

  constexpr size_t SZ_X   = (size_t)PB * PN * PC;    // 4,194,304
  constexpr size_t SZ_QRY = (size_t)PB * PQN * PC;   // 2,097,152
  constexpr size_t SZ_C2  = (size_t)PC * PC;         // 1,048,576
  constexpr size_t BHND   = (size_t)PB * PH * PN * PD;
  constexpr size_t BHQD   = (size_t)PB * PH * PQN * PD;

  bf16* ws = (bf16*)d_ws;
  bf16* xbf      = ws;                 ws += SZ_X;
  bf16* qrybf    = ws;                 ws += SZ_QRY;
  bf16* wqkvbf   = ws;                 ws += 3 * SZ_C2;
  bf16* wkvbf    = ws;                 ws += 2 * SZ_C2;
  bf16* wqlinbf  = ws;                 ws += 3 * SZ_C2;
  bf16* wprojbf  = ws;                 ws += SZ_C2;
  bf16* wqprojbf = ws;                 ws += SZ_C2;
  bf16* qb       = ws;                 ws += BHND;  // [B,H,N,D]
  bf16* kb       = ws;                 ws += BHND;  // [B,H,N,D]
  bf16* vtb      = ws;                 ws += BHND;  // [B,H,D,N]
  bf16* xk0      = ws;                 ws += BHND;  // [B,H,N,D]
  bf16* xv0t     = ws;                 ws += BHND;  // [B,H,D,N]
  bf16* qqb      = ws;                 ws += BHQD;  // [B,H,QN,D]
  bf16* qkb      = ws;                 ws += BHQD;  // [B,H,QN,D]
  bf16* qvt      = ws;                 ws += BHQD;  // [B,H,D,QN]
  bf16* xattnbf  = ws;                 ws += BHND;  // [B,N,H,D]
  bf16* qattnA   = ws;                 ws += BHQD;  // [B,QN,H,D]
  bf16* qattnB   = ws;                 ws += BHQD;  // [B,QN,H,D]
  bf16* qattnS   = ws;                 ws += BHQD;  // [B,QN,H,D]

  dim3 blk256(256), blk128(128);

  // 0) bulk f32 -> bf16 conversions (activations + weights)
  cvt_f32_bf16<<<dim3(SZ_X / 4 / 256), blk256, 0, stream>>>(x, xbf, SZ_X / 4);
  cvt_f32_bf16<<<dim3(SZ_QRY / 4 / 256), blk256, 0, stream>>>(query, qrybf,
                                                              SZ_QRY / 4);
  cvt_f32_bf16<<<dim3(3 * SZ_C2 / 4 / 256), blk256, 0, stream>>>(
      w_qkv, wqkvbf, 3 * SZ_C2 / 4);
  cvt_f32_bf16<<<dim3(2 * SZ_C2 / 4 / 256), blk256, 0, stream>>>(
      w_kv, wkvbf, 2 * SZ_C2 / 4);
  cvt_f32_bf16<<<dim3(3 * SZ_C2 / 4 / 256), blk256, 0, stream>>>(
      w_qlin, wqlinbf, 3 * SZ_C2 / 4);
  cvt_f32_bf16<<<dim3(SZ_C2 / 4 / 256), blk256, 0, stream>>>(w_proj, wprojbf,
                                                             SZ_C2 / 4);
  cvt_f32_bf16<<<dim3(SZ_C2 / 4 / 256), blk256, 0, stream>>>(w_qproj, wqprojbf,
                                                             SZ_C2 / 4);

  // 1) qkv = x @ w_qkv^T  -> q,k (head-major), v (transposed)
  gemm_bf16_wmma<<<dim3((PB * PN) / 128, (3 * PC) / 128), blk256, 0, stream>>>(
      xbf, wqkvbf, PB * PN, 3 * PC, PC, 0, nullptr, nullptr, qb, kb, vtb, 2, PN);
  // 2) kv = x @ w_kv^T    -> xk0, xv0 (transposed)
  gemm_bf16_wmma<<<dim3((PB * PN) / 128, (2 * PC) / 128), blk256, 0, stream>>>(
      xbf, wkvbf, PB * PN, 2 * PC, PC, 0, nullptr, nullptr, xk0, xv0t, nullptr,
      1, PN);
  // 3) qq/qk/qv = query @ w_qlin^T
  gemm_bf16_wmma<<<dim3((PB * PQN) / 128, (3 * PC) / 128), blk256, 0,
                   stream>>>(qrybf, wqlinbf, PB * PQN, 3 * PC, PC, 0, nullptr,
                             nullptr, qqb, qkb, qvt, 2, PQN);

  // 4) self attention: x_attn = sdpa(q,k,v)
  flash_attn_wmma<<<dim3(PN / 64, PB * PH), blk128, 0, stream>>>(
      qb, kb, vtb, xattnbf, PN, PN, nullptr);
  // 5) cross part A: sdpa(qq, qk, qv)
  flash_attn_wmma<<<dim3(PQN / 64, PB * PH), blk128, 0, stream>>>(
      qqb, qkb, qvt, qattnA, PQN, PQN, nullptr);
  // 6) cross part B: tanh(gate_h) * sdpa(qq, xk0, xv0)
  flash_attn_wmma<<<dim3(PQN / 64, PB * PH), blk128, 0, stream>>>(
      qqb, xk0, xv0t, qattnB, PQN, PN, gate);
  // 6b) combine
  add_bf16_vec<<<dim3(BHQD / 8 / 256), blk256, 0, stream>>>(qattnA, qattnB,
                                                            qattnS, BHQD / 8);

  // 7) x_out = x_attn @ w_proj^T + b_proj
  gemm_bf16_wmma<<<dim3((PB * PN) / 128, PC / 128), blk256, 0, stream>>>(
      xattnbf, wprojbf, PB * PN, PC, PC, 1, x_out, b_proj, nullptr, nullptr,
      nullptr, 0, 0);
  // 8) q_out = q_attn @ w_qproj^T + b_qproj
  gemm_bf16_wmma<<<dim3((PB * PQN) / 128, PC / 128), blk256, 0, stream>>>(
      qattnS, wqprojbf, PB * PQN, PC, PC, 1, q_out, b_qproj, nullptr, nullptr,
      nullptr, 0, 0);

  (void)in_sizes; (void)n_in; (void)out_size; (void)ws_size;
}
